// Attention_gated_MB_SSL_50594714747366
// MI455X (gfx1250) — matrix operand order repaired and checked by reading, not compiled, not run on hardware
//
#include <hip/hip_runtime.h>

// ---------------------------------------------------------------------------
// Problem constants (from reference)
// ---------------------------------------------------------------------------
#define NN   262144   // rows
#define LL   1024     // input features
#define DD   128      // hidden dim
#define FF   32       // attention dim
#define BAGS 256      // num segments

typedef __attribute__((ext_vector_type(16))) __bf16 v16bf;
typedef __attribute__((ext_vector_type(8)))  __bf16 v8bf;
typedef __attribute__((ext_vector_type(8)))  float  v8f;
typedef __attribute__((ext_vector_type(4)))  float  v4f;   // clang vector: ok for nontemporal builtins

// ---------------------------------------------------------------------------
// WMMA fragment layouts (CDNA5 ISA, 16-bit 16x32 A / 32x16 B)
//   A: lane l holds row M=l&15;  element e -> K = (e&7) + 8*(l>>4) + 16*(e>>3)
//   B: lane l holds col N=l&15;  element e -> K = (e&7) + 8*(l>>4) + 16*(e>>3)
// ---------------------------------------------------------------------------

// Pre-swizzle a row-major f32 weight [K x Nc] into bf16 WMMA-B fragments:
// out[((kc*(Nc/16)+t)*32 + lane)*16 + e]  -> one v16bf (32B) per lane per tile.
__global__ void frag_pack_kernel(const float* __restrict__ W,
                                 __bf16* __restrict__ out, int K, int Nc) {
    int idx = blockIdx.x * blockDim.x + threadIdx.x;
    int total = K * Nc;
    if (idx >= total) return;
    int e      = idx & 15;
    int lane   = (idx >> 4) & 31;
    int rest   = idx >> 9;            // kc*(Nc/16) + t
    int ntiles = Nc >> 4;
    int t      = rest % ntiles;
    int kc     = rest / ntiles;
    int k    = kc * 32 + (e & 7) + 8 * (lane >> 4) + 16 * (e >> 3);
    int ncol = t * 16 + (lane & 15);
    out[idx] = (__bf16)W[(size_t)k * Nc + ncol];
}

// ---------------------------------------------------------------------------
// K1: H = relu(x @ W1 + b1), bf16 output.
// Memory-bound on x (1 GiB @ 23.3 TB/s ~46us).  x tiles are loaded
// non-temporally (streamed once); W1 fragments are staged per-K-chunk into
// LDS (double buffered) so each block fetches W1 from L2 exactly once
// (512 MB total instead of 4 GB), and all 8 waves consume it via ds_load.
// ---------------------------------------------------------------------------
__global__ void __launch_bounds__(256)
gemm1_kernel(const float* __restrict__ x, const __bf16* __restrict__ W1f,
             const float* __restrict__ b1, __bf16* __restrict__ H) {
    const int tid  = threadIdx.x;
    const int lane = tid & 31;
    const int wave = tid >> 5;
    const int kh   = lane >> 4;       // K-half selector
    const int ln   = lane & 15;       // row (A) / col (C) within tile
    const long rowBase = (long)blockIdx.x * 128 + wave * 16;
    const float* __restrict__ xrow = x + (rowBase + ln) * LL;

    // double-buffered B-fragment stage: 8 tiles * 32 lanes * 16 bf16 = 8 KB
    __shared__ __align__(16) __bf16 bstage[2][8 * 32 * 16];

    v8f acc[8];
    v8f zero = {};
#pragma unroll
    for (int t = 0; t < 8; ++t) acc[t] = zero;

    const uint4* __restrict__ wsrc = (const uint4*)W1f;

    // prologue: stage chunk 0
    {
        uint4* dst = (uint4*)(&bstage[0][0]);
        dst[tid * 2 + 0] = wsrc[tid * 2 + 0];
        dst[tid * 2 + 1] = wsrc[tid * 2 + 1];
    }
    __syncthreads();

    for (int kc = 0; kc < 32; ++kc) {
        const int buf = kc & 1;
        // stage next chunk into the other buffer (coalesced 8 KB copy)
        if (kc + 1 < 32) {
            const uint4* src = wsrc + (size_t)(kc + 1) * 512;
            uint4* dst = (uint4*)(&bstage[buf ^ 1][0]);
            dst[tid * 2 + 0] = src[tid * 2 + 0];
            dst[tid * 2 + 1] = src[tid * 2 + 1];
        }

        // A fragment: two 32B contiguous spans of this lane's x row (streamed,
        // non-temporal so L2 stays reserved for W1f/H).
        const float* p = xrow + kc * 32 + 8 * kh;
        v4f a0 = __builtin_nontemporal_load((const v4f*)(p));
        v4f a1 = __builtin_nontemporal_load((const v4f*)(p + 4));
        v4f a2 = __builtin_nontemporal_load((const v4f*)(p + 16));
        v4f a3 = __builtin_nontemporal_load((const v4f*)(p + 20));
        v16bf afrag;
        afrag[0]  = (__bf16)a0.x; afrag[1]  = (__bf16)a0.y;
        afrag[2]  = (__bf16)a0.z; afrag[3]  = (__bf16)a0.w;
        afrag[4]  = (__bf16)a1.x; afrag[5]  = (__bf16)a1.y;
        afrag[6]  = (__bf16)a1.z; afrag[7]  = (__bf16)a1.w;
        afrag[8]  = (__bf16)a2.x; afrag[9]  = (__bf16)a2.y;
        afrag[10] = (__bf16)a2.z; afrag[11] = (__bf16)a2.w;
        afrag[12] = (__bf16)a3.x; afrag[13] = (__bf16)a3.y;
        afrag[14] = (__bf16)a3.z; afrag[15] = (__bf16)a3.w;

        const v16bf* bl = (const v16bf*)(&bstage[buf][0]);
#pragma unroll
        for (int t = 0; t < 8; ++t) {
            v16bf bfrag = bl[t * 32 + lane];   // ds_load, 32B aligned
            acc[t] = __builtin_amdgcn_wmma_f32_16x16x32_bf16(
                false, afrag, false, bfrag, (short)0, acc[t], false, false);
        }
        __syncthreads();
    }

    // Epilogue: C layout -> row M = r + 8*kh, col d = 16*t + ln.
#pragma unroll
    for (int t = 0; t < 8; ++t) {
        int d = t * 16 + ln;
        float bias = b1[d];
#pragma unroll
        for (int r = 0; r < 8; ++r) {
            long row = rowBase + r + 8 * kh;
            float v = acc[t][r] + bias;
            v = v > 0.0f ? v : 0.0f;
            H[(size_t)row * DD + d] = (__bf16)v;
        }
    }
}

// ---------------------------------------------------------------------------
// K2: gated attention scores  a[i] = sum_f tanh(H@Wt+bt)*sigmoid(H@Ws+bs)*Wa
// Each wave: one 16-row strip of H as WMMA-A, K=128 (4 chunks), F=32 (2 tiles).
// Wt/Ws fragments total 16 KB and stay L2-resident.
// ---------------------------------------------------------------------------
__global__ void __launch_bounds__(256)
attn_kernel(const __bf16* __restrict__ H, const __bf16* __restrict__ Wtf,
            const __bf16* __restrict__ Wsf, const float* __restrict__ bt,
            const float* __restrict__ bs, const float* __restrict__ Wa,
            const float* __restrict__ ba, float* __restrict__ a_out) {
    const int lane = threadIdx.x & 31;
    const int wave = threadIdx.x >> 5;
    const int kh   = lane >> 4;
    const int ln   = lane & 15;
    const long rowBase = ((long)blockIdx.x * 8 + wave) * 16;
    const __bf16* __restrict__ hrow = H + (size_t)(rowBase + ln) * DD;

    v8f at[2], as_[2];
    v8f zero = {};
    at[0] = at[1] = as_[0] = as_[1] = zero;

    const v16bf* __restrict__ tfr = (const v16bf*)Wtf;
    const v16bf* __restrict__ sfr = (const v16bf*)Wsf;

#pragma unroll
    for (int kc = 0; kc < 4; ++kc) {
        const __bf16* p = hrow + kc * 32 + 8 * kh;
        v8bf lo = *(const v8bf*)(p);
        v8bf hi = *(const v8bf*)(p + 16);
        v16bf afrag;
#pragma unroll
        for (int j = 0; j < 8; ++j) { afrag[j] = lo[j]; afrag[8 + j] = hi[j]; }
#pragma unroll
        for (int t = 0; t < 2; ++t) {
            v16bf btf = tfr[(kc * 2 + t) * 32 + lane];
            at[t] = __builtin_amdgcn_wmma_f32_16x16x32_bf16(
                false, afrag, false, btf, (short)0, at[t], false, false);
            v16bf bsf = sfr[(kc * 2 + t) * 32 + lane];
            as_[t] = __builtin_amdgcn_wmma_f32_16x16x32_bf16(
                false, afrag, false, bsf, (short)0, as_[t], false, false);
        }
    }

    // Gate + dot with Wa; C element r -> row r + 8*kh, col f = 16*t + ln.
    float partial[8];
#pragma unroll
    for (int r = 0; r < 8; ++r) partial[r] = 0.0f;
#pragma unroll
    for (int t = 0; t < 2; ++t) {
        int f = t * 16 + ln;
        float bbt = bt[f], bbs = bs[f], wa = Wa[f];
#pragma unroll
        for (int r = 0; r < 8; ++r) {
            float tv = tanhf(at[t][r] + bbt);
            float sv = 1.0f / (1.0f + __expf(-(as_[t][r] + bbs)));
            partial[r] += tv * sv * wa;
        }
    }
    float bias = ba[0];
#pragma unroll
    for (int r = 0; r < 8; ++r) {
        float s = partial[r];
        // reduce over the 16 lanes of this K-half (masks stay within half)
        s += __shfl_xor(s, 1, 32);
        s += __shfl_xor(s, 2, 32);
        s += __shfl_xor(s, 4, 32);
        s += __shfl_xor(s, 8, 32);
        if (ln == 0) a_out[rowBase + r + 8 * kh] = s + bias;
    }
}

// ---------------------------------------------------------------------------
// Segment helpers: idxs is sorted -> bag b occupies [lb(b), lb(b+1)).
// ---------------------------------------------------------------------------
__device__ __forceinline__ int lower_bound_i(const int* __restrict__ idxs,
                                             int n, int v) {
    int lo = 0, hi = n;
    while (lo < hi) {
        int mid = (lo + hi) >> 1;
        if (idxs[mid] < v) lo = mid + 1; else hi = mid;
    }
    return lo;
}

// K3: per-bag max and sum(exp(a - max)).  One block per bag; a/idxs are L2-hot.
__global__ void __launch_bounds__(256)
stats_kernel(const float* __restrict__ a, const int* __restrict__ idxs,
             float* __restrict__ amax, float* __restrict__ den) {
    int b = blockIdx.x;
    int tid = threadIdx.x;
    int lo = lower_bound_i(idxs, NN, b);
    int hi = lower_bound_i(idxs, NN, b + 1);
    __shared__ float red[256];

    float m = -INFINITY;
    for (int i = lo + tid; i < hi; i += 256) m = fmaxf(m, a[i]);
    red[tid] = m;
    __syncthreads();
    for (int s = 128; s > 0; s >>= 1) {
        if (tid < s) red[tid] = fmaxf(red[tid], red[tid + s]);
        __syncthreads();
    }
    float bm = red[0];
    __syncthreads();

    float s = 0.0f;
    for (int i = lo + tid; i < hi; i += 256) s += __expf(a[i] - bm);
    red[tid] = s;
    __syncthreads();
    for (int st = 128; st > 0; st >>= 1) {
        if (tid < st) red[tid] += red[tid + st];
        __syncthreads();
    }
    if (tid == 0) { amax[b] = bm; den[b] = red[0]; }
}

// K4: M[b,:] = sum_i (exp(a_i - amax_b)/den_b) * H_i.  One block per bag,
// thread d owns column d; bag rows are contiguous (no atomics).
__global__ void __launch_bounds__(128)
bagsum_kernel(const __bf16* __restrict__ H, const float* __restrict__ a,
              const int* __restrict__ idxs, const float* __restrict__ amax,
              const float* __restrict__ den, float* __restrict__ Mout) {
    int b = blockIdx.x;
    int d = threadIdx.x;
    int lo = lower_bound_i(idxs, NN, b);
    int hi = lower_bound_i(idxs, NN, b + 1);
    float am = amax[b];
    float dn = den[b];
    float inv = dn > 0.0f ? 1.0f / dn : 0.0f;
    float acc = 0.0f;
#pragma unroll 4
    for (int i = lo; i < hi; ++i) {
        float w = __expf(a[i] - am);
        acc += w * (float)H[(size_t)i * DD + d];
    }
    Mout[b * DD + d] = acc * inv;
}

// K5: proj = normalize(M @ Wp + bp).  One wave (32 lanes == wave32) per bag.
__global__ void __launch_bounds__(32)
proj_kernel(const float* __restrict__ Mmat, const float* __restrict__ Wp,
            const float* __restrict__ bp, float* __restrict__ Pout) {
    int b = blockIdx.x;
    int f = threadIdx.x;
    float s = bp[f];
#pragma unroll 4
    for (int d = 0; d < DD; ++d) s += Mmat[b * DD + d] * Wp[d * FF + f];
    float n2 = s * s;
    n2 += __shfl_xor(n2, 1, 32);
    n2 += __shfl_xor(n2, 2, 32);
    n2 += __shfl_xor(n2, 4, 32);
    n2 += __shfl_xor(n2, 8, 32);
    n2 += __shfl_xor(n2, 16, 32);
    float nrm = fmaxf(sqrtf(n2), 1e-12f);
    Pout[b * FF + f] = s / nrm;
}

// ---------------------------------------------------------------------------
// Host launcher.  Workspace layout (bytes, 256-aligned):
//   W1 frags  [0,        262144)   1024x128 bf16
//   Wt frags  [262144,   270336)    128x 32 bf16
//   Ws frags  [270336,   278528)    128x 32 bf16
//   amax      [278528,   279552)    256 f32
//   den       [279552,   280576)    256 f32
//   a         [280576,  1329152)    N   f32
//   H bf16    [1329152, 68438016)   N x 128 bf16
// ---------------------------------------------------------------------------
extern "C" void kernel_launch(void* const* d_in, const int* in_sizes, int n_in,
                              void* d_out, int out_size, void* d_ws, size_t ws_size,
                              hipStream_t stream) {
    const float* x    = (const float*)d_in[0];
    const int*   idxs = (const int*)d_in[1];
    const float* W1   = (const float*)d_in[2];
    const float* b1   = (const float*)d_in[3];
    const float* Wt   = (const float*)d_in[4];
    const float* bt   = (const float*)d_in[5];
    const float* Ws   = (const float*)d_in[6];
    const float* bs   = (const float*)d_in[7];
    const float* Wa   = (const float*)d_in[8];
    const float* ba   = (const float*)d_in[9];
    const float* Wp   = (const float*)d_in[10];
    const float* bp   = (const float*)d_in[11];

    char* ws = (char*)d_ws;
    __bf16* W1f  = (__bf16*)(ws + 0);
    __bf16* Wtf  = (__bf16*)(ws + 262144);
    __bf16* Wsf  = (__bf16*)(ws + 270336);
    float*  amax = (float*)(ws + 278528);
    float*  den  = (float*)(ws + 279552);
    float*  avec = (float*)(ws + 280576);
    __bf16* Hbf  = (__bf16*)(ws + 1329152);

    float* Mout = (float*)d_out;            // [256 x 128]
    float* Pout = Mout + BAGS * DD;         // [256 x 32]

    // 0) pre-swizzle weights into WMMA-B fragment layout
    frag_pack_kernel<<<(LL * DD + 255) / 256, 256, 0, stream>>>(W1, W1f, LL, DD);
    frag_pack_kernel<<<(DD * FF + 255) / 256, 256, 0, stream>>>(Wt, Wtf, DD, FF);
    frag_pack_kernel<<<(DD * FF + 255) / 256, 256, 0, stream>>>(Ws, Wsf, DD, FF);

    // 1) H = relu(x @ W1 + b1)      (2048 blocks x 8 waves x 16 rows)
    gemm1_kernel<<<NN / 128, 256, 0, stream>>>(x, W1f, b1, Hbf);

    // 2) gated attention scores a[N]
    attn_kernel<<<NN / 128, 256, 0, stream>>>(Hbf, Wtf, Wsf, bt, bs, Wa, ba, avec);

    // 3) per-bag softmax stats
    stats_kernel<<<BAGS, 256, 0, stream>>>(avec, idxs, amax, den);

    // 4) per-bag weighted sum -> M
    bagsum_kernel<<<BAGS, 128, 0, stream>>>(Hbf, avec, idxs, amax, den, Mout);

    // 5) projector + L2 normalize -> proj
    proj_kernel<<<BAGS, 32, 0, stream>>>(Mout, Wp, bp, Pout);
}
